// HSM3Head_60928406061391
// MI455X (gfx1250) — compile-verified, matrix operand order checked by reading, not run on hardware
//
#include <hip/hip_runtime.h>
#include <math.h>

#define B_ 4
#define F_ 256
#define T_ 256
#define K_ 8
#define C_ 10
#define D_ 156
#define N_ 2
#define NEGV (-1.0e9f)
#define SLOT_PAD 320   // padded stride for the N_*D_=312 alpha slots (64B aligned)

typedef __attribute__((ext_vector_type(16))) _Float16 v16h;
typedef __attribute__((ext_vector_type(4)))  _Float16 v4h;
typedef __attribute__((ext_vector_type(8)))  float    v8f;

#ifndef __has_builtin
#define __has_builtin(x) 0
#endif
#if __has_builtin(__builtin_amdgcn_tensor_load_to_lds) && \
    __has_builtin(__builtin_amdgcn_s_wait_tensorcnt)
#define USE_TDM 1
#else
#define USE_TDM 0
#endif

__device__ __forceinline__ void lse_merge(float& m, float& s, float mo, float so) {
    float mm = fmaxf(m, mo);
    s = s * __expf(m - mm) + so * __expf(mo - mm);
    m = mm;
}
__device__ __forceinline__ float lse2(float x0, float x1) {
    float hi = fmaxf(x0, x1);
    return hi + log1pf(__expf(fminf(x0, x1) - hi));
}
__device__ __forceinline__ float log_sigmoidf(float v) {
    return fminf(v, 0.0f) - log1pf(__expf(-fabsf(v)));
}
__device__ __forceinline__ float softplusf(float v) {
    return (v > 20.0f) ? v : log1pf(__expf(v));
}

// ---------------- BatchNorm statistics: mean/rstd per feature over (B,T) ----------------
__global__ __launch_bounds__(256) void k_bn_stats(const float* __restrict__ h,
                                                  float* __restrict__ mean,
                                                  float* __restrict__ rstd) {
    int f = blockIdx.x, t = threadIdx.x;
    float s1 = 0.f, s2 = 0.f;
    for (int b = 0; b < B_; ++b) {
        float v = h[((size_t)b * F_ + f) * T_ + t];
        s1 += v; s2 += v * v;
    }
    __shared__ float r1[256], r2[256];
    r1[t] = s1; r2[t] = s2; __syncthreads();
    for (int off = 128; off; off >>= 1) {
        if (t < off) { r1[t] += r1[t + off]; r2[t] += r2[t + off]; }
        __syncthreads();
    }
    if (t == 0) {
        float m = r1[0] / (float)(B_ * T_);
        float var = r2[0] / (float)(B_ * T_) - m * m;
        mean[f] = m;
        rstd[f] = rsqrtf(var + 1e-5f);
    }
}

// ---------------- normalize + relu -> xnorm ----------------
__global__ __launch_bounds__(256) void k_bn_apply(const float* __restrict__ h,
                                                  const float* __restrict__ mean,
                                                  const float* __restrict__ rstd,
                                                  const float* __restrict__ gamma,
                                                  const float* __restrict__ beta,
                                                  float* __restrict__ xn) {
    int idx = blockIdx.x * 256 + threadIdx.x;
    int f = (idx >> 8) & (F_ - 1);
    float v = (h[idx] - mean[f]) * rstd[f] * gamma[f] + beta[f];
    xn[idx] = fmaxf(v, 0.0f);
}

#if USE_TDM
typedef __attribute__((ext_vector_type(4))) unsigned int v4u_t;
typedef __attribute__((ext_vector_type(8))) int v8i_t;
typedef __attribute__((ext_vector_type(4))) int v4i_t;

// Issue a TDM 2D tile load (f32 elements): tile_rows x tile_cols from a row-major
// tensor with row stride row_stride_elems, into LDS at byte offset lds_off.
__device__ __forceinline__ void tdm_load_f32_tile(unsigned int lds_off,
                                                  const float* gptr,
                                                  int tile_cols, int tile_rows,
                                                  int row_stride_elems) {
    unsigned long long ga = (unsigned long long)(size_t)gptr;
    v4u_t g0;
    g0[0] = 1u;                                                        // count=1, user D#
    g0[1] = lds_off;                                                   // lds_addr (bytes)
    g0[2] = (unsigned int)ga;                                          // global_addr[31:0]
    g0[3] = (unsigned int)((ga >> 32) & 0x01ffffffull) | (2u << 30);   // addr[56:32] | type=2
    v8i_t g1;
    g1[0] = 0x00020000;                                      // data_size=4B, mask/pad/iter=0
    g1[1] = (row_stride_elems & 0xffff) << 16;               // tensor_dim0 lo16
    g1[2] = ((row_stride_elems >> 16) & 0xffff)              // tensor_dim0 hi16
          | ((256 & 0xffff) << 16);                          // tensor_dim1 lo16
    g1[3] = (tile_cols & 0xffff) << 16;                      // tensor_dim1 hi16=0 | tile_dim0
    g1[4] = (tile_rows & 0xffff);                            // tile_dim1 (tile_dim2=0)
    g1[5] = row_stride_elems;                                // tensor_dim0_stride lo32
    g1[6] = 0;                                               // stride hi16 | dim1_stride=0
    g1[7] = 0;
    v4i_t zz = {0, 0, 0, 0};
#if defined(__clang_major__) && (__clang_major__ >= 23)
    v8i_t z8 = {0, 0, 0, 0, 0, 0, 0, 0};
    __builtin_amdgcn_tensor_load_to_lds(g0, g1, zz, zz, z8, 0);
#else
    __builtin_amdgcn_tensor_load_to_lds(g0, g1, zz, zz, 0);
#endif
}
#endif  // USE_TDM

// ---------------- WMMA f16 GEMM: Y[b,o,t] = sum_f W[o,f]*X[b,f,t] + bias[o] ----------------
// grid: (M/16, T/64, B); block 128 (4 waves); wave w owns the 16x16 tile at t-offset w*16.
// Operands are staged in LDS directly in WMMA register layout (lane-major), so the
// compute side does 2x ds_load_b128 per operand instead of 16x ds_load_u16.
// The X tile is DMA'd into LDS by the Tensor Data Mover, double-buffered, one
// iteration ahead (wave 0 issues, s_wait_tensorcnt synchronizes).
__global__ __launch_bounds__(128) void k_wmma_gemm(const float* __restrict__ W,
                                                   const float* __restrict__ X,
                                                   const float* __restrict__ bias,
                                                   float* __restrict__ Y,
                                                   int M, int do_relu) {
    int b  = blockIdx.z;
    int o0 = blockIdx.x * 16;
    int t0 = blockIdx.y * 64;
    int tid = threadIdx.x;
    int wave = tid >> 5, lane = tid & 31;

    __shared__ __align__(32) _Float16 AsL[32 * 16];       // [lane][16]  A operand
    __shared__ __align__(32) _Float16 BsL[4 * 32 * 16];   // [wave][lane][16]  B operand
#if USE_TDM
    __shared__ __align__(16) float XsF[2][32 * 64];       // TDM landing pads (row-major f32)
#endif

    v8f acc = {0.f, 0.f, 0.f, 0.f, 0.f, 0.f, 0.f, 0.f};
    const float* Xb = X + (size_t)b * F_ * T_;

#if USE_TDM
    if (tid < 32)
        tdm_load_f32_tile((unsigned int)(size_t)&XsF[0][0], Xb + t0, 64, 32, T_);
#endif

#pragma unroll
    for (int it = 0; it < 8; ++it) {
        int f0 = it * 32;
#if USE_TDM
        if (tid < 32) {
            if (it < 7) {
                tdm_load_f32_tile((unsigned int)(size_t)&XsF[(it + 1) & 1][0],
                                  Xb + (size_t)(f0 + 32) * T_ + t0, 64, 32, T_);
                __builtin_amdgcn_s_wait_tensorcnt(1);   // current buffer complete
            } else {
                __builtin_amdgcn_s_wait_tensorcnt(0);
            }
        }
#endif
        __syncthreads();

        // ---- stage A: 16x32 f32 tile -> lane-major f16 operand (1 b128 load + 1 b64 store)
        {
            int m = tid >> 3, q = tid & 7;               // quad q covers K = 4q..4q+3
            const float4 w4 = *(const float4*)&W[(size_t)(o0 + m) * F_ + f0 + 4 * q];
            int hi = (q >> 1) & 1;
            int j0 = 2 * (q & 1) + ((q >= 4) ? 4 : 0);
            int la = m + 16 * hi;
            v4h pk;
            pk[0] = (_Float16)w4.x; pk[1] = (_Float16)w4.y;
            pk[2] = (_Float16)w4.z; pk[3] = (_Float16)w4.w;
            *(v4h*)&AsL[la * 16 + 2 * j0] = pk;
        }
        // ---- stage B: 32x64 tile -> lane-major f16 operand per wave
#pragma unroll
        for (int it2 = 0; it2 < 4; ++it2) {
            int w = it2 * 128 + tid;
            int tl = w & 63, q = w >> 6;                 // quad q covers K = 4q..4q+3
            float x0, x1, x2, x3;
#if USE_TDM
            const float* src = &XsF[it & 1][4 * q * 64 + tl];
            x0 = src[0]; x1 = src[64]; x2 = src[128]; x3 = src[192];
#else
            const float* src = Xb + (size_t)(f0 + 4 * q) * T_ + t0 + tl;
            x0 = src[0]; x1 = src[T_]; x2 = src[2 * T_]; x3 = src[3 * T_];
#endif
            int hi = (q >> 1) & 1;
            int j0 = 2 * (q & 1) + ((q >= 4) ? 4 : 0);
            int lb = (tl & 15) + 16 * hi;
            int wg = tl >> 4;
            v4h pk;
            pk[0] = (_Float16)x0; pk[1] = (_Float16)x1;
            pk[2] = (_Float16)x2; pk[3] = (_Float16)x3;
            *(v4h*)&BsL[(wg * 32 + lb) * 16 + 2 * j0] = pk;
        }
        __syncthreads();

        v16h av = *(const v16h*)&AsL[lane * 16];
        v16h bv = *(const v16h*)&BsL[(wave * 32 + lane) * 16];
        acc = __builtin_amdgcn_wmma_f32_16x16x32_f16(false, av, false, bv,
                                                     (short)0, acc, false, false);
        __syncthreads();
    }

    int chh = lane >> 4, cn = lane & 15;
#pragma unroll
    for (int r = 0; r < 8; ++r) {
        int mrow = r + 8 * chh;                 // C/D layout: VGPR r -> rows r and r+8
        int o = o0 + mrow;
        float v = acc[r] + bias[o];
        if (do_relu) v = fmaxf(v, 0.0f);
        Y[((size_t)b * M + o) * T_ + t0 + wave * 16 + cn] = v;
    }
}

// ---------------- attention pooling -> log_pi (B,K) ----------------
__global__ __launch_bounds__(256) void k_pool(const float* __restrict__ xm,
                                              const float* __restrict__ q,
                                              const float* __restrict__ wat,
                                              const float* __restrict__ bat,
                                              const float* __restrict__ wk,
                                              const float* __restrict__ bk,
                                              float* __restrict__ log_pi) {
    int b = blockIdx.x, tid = threadIdx.x, lane = tid & 31, wv = tid >> 5;
    __shared__ float wbuf[256], pbuf[256], zbuf[256];
    __shared__ float pm[8], ps[8];
    __shared__ float lbuf[K_];
    const float* X = xm + (size_t)b * F_ * T_;

    float sc = 0.f;
    for (int f = 0; f < F_; ++f) sc += q[f] * X[(size_t)f * T_ + tid];

    float m = sc, s = 1.f;
    for (int off = 16; off; off >>= 1)
        lse_merge(m, s, __shfl_xor(m, off, 32), __shfl_xor(s, off, 32));
    if (lane == 0) { pm[wv] = m; ps[wv] = s; }
    __syncthreads();
    if (tid == 0) {
        float mm = pm[0], ss = ps[0];
        for (int w = 1; w < 8; ++w) lse_merge(mm, ss, pm[w], ps[w]);
        pm[0] = mm; ps[0] = ss;
    }
    __syncthreads();
    wbuf[tid] = __expf(sc - pm[0]) / ps[0];
    __syncthreads();

    float pv = 0.f;
    for (int t = 0; t < T_; ++t) pv += X[(size_t)tid * T_ + t] * wbuf[t];
    pbuf[tid] = pv;
    __syncthreads();

    float z = bat[tid];
    for (int f2 = 0; f2 < F_; ++f2) z += pbuf[f2] * wat[(size_t)tid * F_ + f2];
    zbuf[tid] = z;
    __syncthreads();

    if (tid < K_) {
        float lg = bk[tid];
        for (int f2 = 0; f2 < F_; ++f2) lg += zbuf[f2] * wk[(size_t)tid * F_ + f2];
        lbuf[tid] = lg;
    }
    __syncthreads();
    if (tid == 0) {
        float mm = lbuf[0];
        for (int k = 1; k < K_; ++k) mm = fmaxf(mm, lbuf[k]);
        float ss = 0.f;
        for (int k = 0; k < K_; ++k) ss += __expf(lbuf[k] - mm);
        float lz = mm + __logf(ss);
        for (int k = 0; k < K_; ++k) log_pi[b * K_ + k] = lbuf[k] - lz;
    }
}

// ---------------- gamma-duration log-pmf (K,C,N,D), log-softmax over D ----------------
__global__ __launch_bounds__(256) void k_gamma(const float* __restrict__ conc,
                                               const float* __restrict__ rate,
                                               float* __restrict__ logpd) {
    int g = blockIdx.x;       // (k*C+c)*N+n, L==1 so mixture log-weight is 0
    int tid = threadIdx.x, lane = tid & 31, wv = tid >> 5;
    __shared__ float sa, sbv, slg;
    __shared__ float pm[8], ps[8];
    __shared__ float lsev;
    if (tid == 0) {
        float a  = softplusf(conc[g]) + 1e-4f;
        float bb = softplusf(rate[g]) + 1e-4f;
        sa = a; sbv = bb; slg = lgammaf(a);
    }
    __syncthreads();
    float lp = NEGV;
    if (tid < D_) {
        float dd = (float)(tid + 1);
        lp = sa * __logf(sbv) - slg + (sa - 1.0f) * __logf(dd) - sbv * dd;
    }
    float m = lp, s = 1.f;
    for (int off = 16; off; off >>= 1)
        lse_merge(m, s, __shfl_xor(m, off, 32), __shfl_xor(s, off, 32));
    if (lane == 0) { pm[wv] = m; ps[wv] = s; }
    __syncthreads();
    if (tid == 0) {
        float mm = pm[0], ss = ps[0];
        for (int w = 1; w < 8; ++w) lse_merge(mm, ss, pm[w], ps[w]);
        lsev = mm + __logf(ss);
    }
    __syncthreads();
    if (tid < D_) logpd[(size_t)g * D_ + tid] = lp - lsev;
}

// ---------------- log-sigmoid pair -> log_lo (K,C,N) ----------------
__global__ __launch_bounds__(128) void k_loglo(const float* __restrict__ lo1,
                                               float* __restrict__ loglo) {
    int i = threadIdx.x;
    if (i < K_ * C_) {
        float v = lo1[i];
        loglo[i * 2 + 0] = log_sigmoidf(v);
        loglo[i * 2 + 1] = log_sigmoidf(-v);
    }
}

// ---------------- HSMM forward: one block (10 waves, 320 thr) per (b,k,c) chain ----------------
// thread tid = n*160 + d; log_b reconstructed from prefix sums P (never materialized).
// alpha stored t-major: alphaS[chain][t][slot=n*156+d], slab stride SLOT_PAD.
__global__ __launch_bounds__(320) void k_forward(const float* __restrict__ logpx,
                                                 const float* __restrict__ logpd,
                                                 const float* __restrict__ loglo,
                                                 const float* __restrict__ logA,
                                                 float* __restrict__ alphaS,
                                                 float* __restrict__ Zout) {
    int chain = blockIdx.x;
    int c = chain % C_, k = (chain / C_) % K_, b = chain / (C_ * K_);
    int tid = threadIdx.x, lane = tid & 31, wv = tid >> 5;
    int n = tid / 160, d = tid % 160;
    bool active = d < D_;
    int kc = k * C_ + c;

    float pd  = active ? logpd[((size_t)(kc * N_ + n)) * D_ + d] : 0.f;
    float lon = loglo[kc * N_ + n];
    float A0n = logA[(kc * N_ + 0) * N_ + n];
    float A1n = logA[(kc * N_ + 1) * N_ + n];

    __shared__ float sP[2][T_];   // inclusive prefix sums of logp_x per state
    __shared__ float sS[2][T_];   // s_t history
    __shared__ float pm[10], ps[10];

    if (tid < T_) {
        int base0 = ((b * 160) + k * 20 + c * 2) * T_;
        sP[0][tid] = logpx[base0 + tid];
        sP[1][tid] = logpx[base0 + T_ + tid];
    }
    __syncthreads();
    for (int off = 1; off < T_; off <<= 1) {       // Hillis-Steele scan
        float v0 = 0.f, v1 = 0.f;
        if (tid < T_ && tid >= off) { v0 = sP[0][tid - off]; v1 = sP[1][tid - off]; }
        __syncthreads();
        if (tid < T_ && tid >= off) { sP[0][tid] += v0; sP[1][tid] += v1; }
        __syncthreads();
    }

    float* aBase = alphaS + (size_t)chain * T_ * SLOT_PAD;
    for (int t = 0; t < T_; ++t) {
        float a;
        if (!active || d > t) a = NEGV;
        else if (d == t)      a = lon + pd + sP[n][t];
        else {
            float tr = lse2(sS[0][t - d - 1] + A0n, sS[1][t - d - 1] + A1n);
            a = tr + pd + (sP[n][t] - sP[n][t - d - 1]);
        }
        if (active) aBase[(size_t)t * SLOT_PAD + n * D_ + d] = a;

        float m = a, s = 1.f;
        for (int off = 16; off; off >>= 1)
            lse_merge(m, s, __shfl_xor(m, off, 32), __shfl_xor(s, off, 32));
        if (lane == 0) { pm[wv] = m; ps[wv] = s; }
        __syncthreads();
        if (tid == 0 || tid == 160) {
            int base = (tid == 0) ? 0 : 5;
            float mm = pm[base], ss = ps[base];
            for (int w = 1; w < 5; ++w) lse_merge(mm, ss, pm[base + w], ps[base + w]);
            sS[n][t] = mm + __logf(ss);
        }
        __syncthreads();
    }
    if (tid == 0) Zout[chain] = lse2(sS[0][T_ - 1], sS[1][T_ - 1]);
}

// ---------------- HSMM backward: one block per chain; betas kept in LDS ----------------
__global__ __launch_bounds__(320) void k_backward(const float* __restrict__ logpx,
                                                  const float* __restrict__ logpd,
                                                  const float* __restrict__ logA,
                                                  float* __restrict__ betaS) {
    int chain = blockIdx.x;
    int c = chain % C_, k = (chain / C_) % K_, b = chain / (C_ * K_);
    int tid = threadIdx.x, lane = tid & 31, wv = tid >> 5;
    int n = tid / 160, j = tid % 160;
    bool active = j < (D_ - 1);
    int kc = k * C_ + c;

    float pdj = active ? logpd[((size_t)(kc * N_ + n)) * D_ + j] : 0.f;
    float Ai0 = 0.f, Ai1 = 0.f;
    if (tid < 2) { Ai0 = logA[(kc * N_ + tid) * N_ + 0]; Ai1 = logA[(kc * N_ + tid) * N_ + 1]; }

    __shared__ float sP[2][T_];
    __shared__ float sB[2][T_];
    __shared__ float pm[10], ps[10];
    __shared__ float sV[2];

    if (tid < T_) {
        int base0 = ((b * 160) + k * 20 + c * 2) * T_;
        sP[0][tid] = logpx[base0 + tid];
        sP[1][tid] = logpx[base0 + T_ + tid];
    }
    if (tid < 2) sB[tid][T_ - 1] = 0.0f;
    __syncthreads();
    for (int off = 1; off < T_; off <<= 1) {
        float v0 = 0.f, v1 = 0.f;
        if (tid < T_ && tid >= off) { v0 = sP[0][tid - off]; v1 = sP[1][tid - off]; }
        __syncthreads();
        if (tid < T_ && tid >= off) { sP[0][tid] += v0; sP[1][tid] += v1; }
        __syncthreads();
    }

    for (int t = T_ - 2; t >= 0; --t) {
        int tau = t + 1 + j;
        float v;
        if (!active || tau > T_ - 1) v = NEGV;
        else v = (sP[n][tau] - sP[n][t]) + sB[n][tau] + pdj;

        float m = v, s = 1.f;
        for (int off = 16; off; off >>= 1)
            lse_merge(m, s, __shfl_xor(m, off, 32), __shfl_xor(s, off, 32));
        if (lane == 0) { pm[wv] = m; ps[wv] = s; }
        __syncthreads();
        if (tid == 0 || tid == 160) {
            int base = (tid == 0) ? 0 : 5;
            float mm = pm[base], ss = ps[base];
            for (int w = 1; w < 5; ++w) lse_merge(mm, ss, pm[base + w], ps[base + w]);
            sV[n] = mm + __logf(ss);
        }
        __syncthreads();
        if (tid < 2) sB[tid][t] = lse2(Ai0 + sV[0], Ai1 + sV[1]);
        __syncthreads();
    }
    if (tid < T_) {
        betaS[((size_t)chain * 2 + 0) * T_ + tid] = sB[0][tid];
        betaS[((size_t)chain * 2 + 1) * T_ + tid] = sB[1][tid];
    }
}

// ---------------- logp_event: LSE over K, LDS tile transpose (t-major -> slot-major) ----------------
__global__ __launch_bounds__(256) void k_event(const float* __restrict__ alphaS,
                                               const float* __restrict__ betaS,
                                               const float* __restrict__ log_pi,
                                               const float* __restrict__ Zws,
                                               float* __restrict__ out) {
    int st = blockIdx.x;             // slot tile (0..9)
    int tt = blockIdx.y;             // t tile (0..7)
    int bc = blockIdx.z;             // b*C + c
    int b = bc / C_, c = bc % C_;
    int tid = threadIdx.x;
    int si = tid & 31, tg = tid >> 5;
    int slot = st * 32 + si;
    bool vs = slot < N_ * D_;
    int n = (slot >= D_) ? 1 : 0;

    float m[4], s[4];
#pragma unroll
    for (int r = 0; r < 4; ++r) { m[r] = -3.4e38f; s[r] = 0.f; }

    for (int k = 0; k < K_; ++k) {
        int chain = (b * K_ + k) * C_ + c;
        const float* aB = alphaS + (size_t)chain * T_ * SLOT_PAD;
        const float* bB = betaS + ((size_t)chain * 2 + n) * T_;
        float extra = log_pi[b * K_ + k] - Zws[chain];
        if (k + 1 < K_) {
            const float* nB = alphaS + (size_t)((b * K_ + k + 1) * C_ + c) * T_ * SLOT_PAD;
            __builtin_prefetch(nB + (size_t)(tt * 32 + tg) * SLOT_PAD + slot, 0, 1);
        }
#pragma unroll
        for (int r = 0; r < 4; ++r) {
            int t = tt * 32 + tg + 8 * r;
            float av = vs ? aB[(size_t)t * SLOT_PAD + slot] : NEGV;
            float val = av + extra + bB[t];
            float mm = fmaxf(m[r], val);
            s[r] = s[r] * __expf(m[r] - mm) + __expf(val - mm);
            m[r] = mm;
        }
    }
    __shared__ float tile[32][33];
#pragma unroll
    for (int r = 0; r < 4; ++r) tile[si][tg + 8 * r] = m[r] + __logf(s[r]);
    __syncthreads();

    int ti = tid & 31, sg = tid >> 5;
#pragma unroll
    for (int r = 0; r < 4; ++r) {
        int sl = st * 32 + sg + 8 * r;
        if (sl < N_ * D_) {
            int no = (sl >= D_) ? 1 : 0;
            int dd = sl - no * D_;
            out[((((size_t)b * C_ + c) * N_ + no) * D_ + dd) * T_ + tt * 32 + ti] =
                tile[sg + 8 * r][ti];
        }
    }
}

// ---------------- p_frame: masked LSE over (d,s) per (b,c,n,t) ----------------
__global__ __launch_bounds__(256) void k_pframe(const float* __restrict__ ev,
                                                float* __restrict__ lj) {
    int t = blockIdx.x, cn = blockIdx.y, b = blockIdx.z;
    int c = cn >> 1, n = cn & 1;
    int tid = threadIdx.x, lane = tid & 31, wv = tid >> 5;
    const float* base = ev + ((((size_t)b * C_ + c) * N_ + n) * D_) * T_;

    float m = -3.4e38f, s = 0.f;
    for (int d = 0; d < D_; ++d) {
        int lo = max(d, t);
        int hi = min(t + d, T_ - 1);
        int sidx = lo + tid;                  // span <= d+1 <= 156 <= 256: <=1 elem/thread
        if (sidx <= hi) {
            float v = base[(size_t)d * T_ + sidx];
            float mm = fmaxf(m, v);
            s = s * __expf(m - mm) + __expf(v - mm);
            m = mm;
        }
    }
    for (int off = 16; off; off >>= 1)
        lse_merge(m, s, __shfl_xor(m, off, 32), __shfl_xor(s, off, 32));
    __shared__ float pm[8], ps[8];
    if (lane == 0) { pm[wv] = m; ps[wv] = s; }
    __syncthreads();
    if (tid == 0) {
        float mm = pm[0], ss = ps[0];
        for (int w = 1; w < 8; ++w) lse_merge(mm, ss, pm[w], ps[w]);
        lj[(((size_t)b * C_ + c) * N_ + n) * T_ + t] = mm + __logf(ss);
    }
}

// ---------------- softmax over N (=2): p_frame = sigmoid(l1 - l0) ----------------
__global__ __launch_bounds__(256) void k_pframe_out(const float* __restrict__ lj,
                                                    float* __restrict__ out2) {
    int idx = blockIdx.x * 256 + threadIdx.x;
    if (idx >= B_ * C_ * T_) return;
    int t = idx % T_;
    int c = (idx / T_) % C_;
    int b = idx / (T_ * C_);
    size_t base = (((size_t)b * C_ + c) * N_) * T_;
    float l0 = lj[base + t], l1 = lj[base + T_ + t];
    out2[idx] = 1.0f / (1.0f + __expf(l0 - l1));
}

extern "C" void kernel_launch(void* const* d_in, const int* in_sizes, int n_in,
                              void* d_out, int out_size, void* d_ws, size_t ws_size,
                              hipStream_t stream) {
    const float* h         = (const float*)d_in[0];
    const float* logit_lo1 = (const float*)d_in[1];
    const float* logA      = (const float*)d_in[2];
    const float* dur_conc  = (const float*)d_in[3];
    const float* dur_rate  = (const float*)d_in[4];
    /* d_in[5] dur_mix unused: L==1 => log_softmax == 0 */
    const float* w_mix     = (const float*)d_in[6];
    const float* b_mix     = (const float*)d_in[7];
    const float* q_attn    = (const float*)d_in[8];
    const float* w_attn    = (const float*)d_in[9];
    const float* b_attn    = (const float*)d_in[10];
    const float* w_k       = (const float*)d_in[11];
    const float* b_k       = (const float*)d_in[12];
    const float* bn_gamma  = (const float*)d_in[13];
    const float* bn_beta   = (const float*)d_in[14];
    const float* w_em      = (const float*)d_in[15];
    const float* b_em      = (const float*)d_in[16];

    float* ws = (float*)d_ws;
    size_t o = 0;
    float* MEAN  = ws + o; o += F_;
    float* RSTD  = ws + o; o += F_;
    float* XNORM = ws + o; o += (size_t)B_ * F_ * T_;
    float* XMIX  = ws + o; o += (size_t)B_ * F_ * T_;
    float* LOGPX = ws + o; o += (size_t)B_ * K_ * C_ * N_ * T_;
    float* LOGPD = ws + o; o += (size_t)K_ * C_ * N_ * D_;
    float* LOGLO = ws + o; o += K_ * C_ * N_;
    float* LOGPI = ws + o; o += B_ * K_;
    float* ZWS   = ws + o; o += B_ * K_ * C_;
    float* BETA  = ws + o; o += (size_t)B_ * K_ * C_ * N_ * T_;
    float* LJ    = ws + o; o += (size_t)B_ * C_ * N_ * T_;
    float* ALPHA = ws + o; o += (size_t)B_ * K_ * C_ * T_ * SLOT_PAD;

    float* OUT_EV = (float*)d_out;
    float* OUT_PF = OUT_EV + (size_t)B_ * C_ * N_ * D_ * T_;

    k_bn_stats<<<dim3(F_), dim3(256), 0, stream>>>(h, MEAN, RSTD);
    k_bn_apply<<<dim3((B_ * F_ * T_) / 256), dim3(256), 0, stream>>>(h, MEAN, RSTD,
                                                                     bn_gamma, bn_beta, XNORM);
    k_wmma_gemm<<<dim3(160 / 16, T_ / 64, B_), dim3(128), 0, stream>>>(w_em, XNORM, b_em,
                                                                       LOGPX, 160, 0);
    k_wmma_gemm<<<dim3(F_ / 16, T_ / 64, B_), dim3(128), 0, stream>>>(w_mix, h, b_mix,
                                                                      XMIX, F_, 1);
    k_pool<<<dim3(B_), dim3(256), 0, stream>>>(XMIX, q_attn, w_attn, b_attn, w_k, b_k, LOGPI);
    k_gamma<<<dim3(K_ * C_ * N_), dim3(256), 0, stream>>>(dur_conc, dur_rate, LOGPD);
    k_loglo<<<dim3(1), dim3(128), 0, stream>>>(logit_lo1, LOGLO);
    k_forward<<<dim3(B_ * K_ * C_), dim3(320), 0, stream>>>(LOGPX, LOGPD, LOGLO, logA,
                                                            ALPHA, ZWS);
    k_backward<<<dim3(B_ * K_ * C_), dim3(320), 0, stream>>>(LOGPX, LOGPD, logA, BETA);
    k_event<<<dim3(10, 8, B_ * C_), dim3(256), 0, stream>>>(ALPHA, BETA, LOGPI, ZWS, OUT_EV);
    k_pframe<<<dim3(T_, C_ * N_, B_), dim3(256), 0, stream>>>(OUT_EV, LJ);
    k_pframe_out<<<dim3((B_ * C_ * T_ + 255) / 256), dim3(256), 0, stream>>>(LJ, OUT_PF);

    (void)in_sizes; (void)n_in; (void)out_size; (void)ws_size;
}
